// RetNetBlock_84310208021099
// MI455X (gfx1250) — matrix operand order repaired and checked by reading, not compiled, not run on hardware
//
#include <hip/hip_runtime.h>
#include <hip/hip_bf16.h>
#include <math.h>

typedef __bf16 bf16;
typedef __attribute__((ext_vector_type(8)))  __bf16 bf16x8;
typedef __attribute__((ext_vector_type(16))) __bf16 bf16x16;
typedef __attribute__((ext_vector_type(8)))  float  f32x8;
typedef __attribute__((ext_vector_type(4)))  unsigned int u32x4;
typedef __attribute__((ext_vector_type(4)))  int i32x4;
typedef __attribute__((ext_vector_type(8)))  int i32x8;

// ---------- helpers ----------
static __device__ __forceinline__ bf16 f2bf(float f) {
  union { float f; unsigned u; } a; a.f = f;
  unsigned r = a.u + 0x7FFFu + ((a.u >> 16) & 1u);   // round-to-nearest-even
  unsigned short h = (unsigned short)(r >> 16);
  bf16 b; __builtin_memcpy(&b, &h, 2);
  return b;
}

static __device__ __forceinline__ float blk_sum(float v, float* buf) {
  const int t = threadIdx.x;
  buf[t] = v; __syncthreads();
  for (int s = 128; s > 0; s >>= 1) {
    if (t < s) buf[t] += buf[t + s];
    __syncthreads();
  }
  float r = buf[0]; __syncthreads();
  return r;
}

static __device__ __forceinline__ float blk_max(float v, float* buf) {
  const int t = threadIdx.x;
  buf[t] = v; __syncthreads();
  for (int s = 128; s > 0; s >>= 1) {
    if (t < s) buf[t] = fmaxf(buf[t], buf[t + s]);
    __syncthreads();
  }
  float r = buf[0]; __syncthreads();
  return r;
}

// ---------- TDM: DMA a 128-row x 32-col bf16 tile (row stride = strideElems)
// into LDS with 16B padding per 64B row (LDS row pitch = 80B = 40 bf16). ----------
static __device__ __forceinline__ void tdm_load_tile(const bf16* g, unsigned lds_addr,
                                                     int strideElems)
{
  const unsigned long long ga = (unsigned long long)(size_t)g;
  u32x4 g0;
  g0[0] = 1u;                                          // count=1 (valid user D#)
  g0[1] = lds_addr;                                    // LDS byte address
  g0[2] = (unsigned)ga;                                // global_addr[31:0]
  g0[3] = ((unsigned)(ga >> 32) & 0x01FFFFFFu) | (2u << 30); // addr[56:32] | type=2
  i32x8 g1;
  // data_size=2B (code 1), pad_enable, pad_interval=16DW(64B) code 3, pad_amount=4DW(16B) code 3
  g1[0] = (int)((1u << 16) | (1u << 20) | (3u << 22) | (3u << 25));
  g1[1] = (int)(32u << 16);        // tensor_dim0[15:0]=32   (bits 79:48)
  g1[2] = (int)(128u << 16);       // tensor_dim0 hi=0 | tensor_dim1[15:0]=128
  g1[3] = (int)(32u << 16);        // tensor_dim1 hi=0 | tile_dim0=32
  g1[4] = (int)128;                // tile_dim1=128 | tile_dim2=0
  g1[5] = strideElems;             // tensor_dim0_stride[31:0]
  g1[6] = 0;                       // stride hi | tensor_dim1_stride lo (unused, 2D)
  g1[7] = 0;
  i32x4 z; z[0] = 0; z[1] = 0; z[2] = 0; z[3] = 0;
#if defined(__clang_major__) && (__clang_major__ >= 23)
  i32x8 z8; for (int i = 0; i < 8; ++i) z8[i] = 0;
  __builtin_amdgcn_tensor_load_to_lds(g0, g1, z, z, z8, 0);
#else
  __builtin_amdgcn_tensor_load_to_lds(g0, g1, z, z, 0);
#endif
}

// ---------- one 32-deep MMA step on a staged LDS tile pair ----------
static __device__ __forceinline__ void mma_step(
    const bf16 (*__restrict__ At)[40], const bf16 (*__restrict__ Bt)[40],
    f32x8 acc[2][4], int wr, int wc, int l15, int lh)
{
  bf16x16 afr[2], bfr[4];
  const int ab = lh * 8;                 // A frag: K {ab..ab+7} and {16+ab..23+ab}
#pragma unroll
  for (int tm = 0; tm < 2; ++tm) {
    const int r = wr * 32 + tm * 16 + l15;
    bf16x8 lo = *(const bf16x8*)&At[r][ab];
    bf16x8 hi = *(const bf16x8*)&At[r][16 + ab];
    afr[tm] = __builtin_shufflevector(lo, hi, 0,1,2,3,4,5,6,7,8,9,10,11,12,13,14,15);
  }
  const int ko = lh * 8;                 // B frag: N=l15, K {ko..ko+7} and {16+ko..23+ko}
#pragma unroll
  for (int tn = 0; tn < 4; ++tn) {
    const int c = wc * 64 + tn * 16 + l15;
    bf16x8 lo = *(const bf16x8*)&Bt[c][ko];
    bf16x8 hi = *(const bf16x8*)&Bt[c][16 + ko];
    bfr[tn] = __builtin_shufflevector(lo, hi, 0,1,2,3,4,5,6,7,8,9,10,11,12,13,14,15);
  }
#pragma unroll
  for (int tm = 0; tm < 2; ++tm)
#pragma unroll
    for (int tn = 0; tn < 4; ++tn)
      acc[tm][tn] = __builtin_amdgcn_wmma_f32_16x16x32_bf16(
          false, afr[tm], false, bfr[tn], (short)0, acc[tm][tn], false, false);
}

// ---------- generic bf16 WMMA GEMM: C = A * op(B) (+bias)(+silu)(+resid) ----------
// A: [M,K] row-major bf16.  TRANS_B: B is [N,K] row-major (TDM double-buffered path);
// else B is [K,N] row-major (sync transpose-staging path).
// 256 threads = 8 waves; block tile 128x128, K-step 32; wave tile 32x64 (2x4 WMMA).
template<bool TRANS_B, bool BIAS, bool RESID, bool SILU, bool OUT_BF16>
__global__ __launch_bounds__(256)
void gemm_wmma(const bf16* __restrict__ A, const bf16* __restrict__ B,
               const float* __restrict__ bias, const float* __restrict__ resid,
               float* __restrict__ outF, bf16* __restrict__ outB,
               int M, int N, int K, long sAb, long sBb, long sCb)
{
  constexpr int NBUF = TRANS_B ? 2 : 1;
  __shared__ alignas(16) bf16 As[NBUF][128][40];
  __shared__ alignas(16) bf16 Bs[NBUF][128][40];

  const int tid  = threadIdx.x;
  const int lane = tid & 31;
  const int wid  = tid >> 5;
  const int wr   = wid >> 1;   // wave row 0..3 (32 rows each)
  const int wc   = wid & 1;    // wave col 0..1 (64 cols each)
  const int l15  = lane & 15;
  const int lh   = lane >> 4;  // 0/1 (lane half)

  const long bm = (long)blockIdx.y * 128;
  const long bn = (long)blockIdx.x * 128;
  const bf16* Ab = A + (long)blockIdx.z * sAb;
  const bf16* Bb = B + (long)blockIdx.z * sBb;
  const long  cb = (long)blockIdx.z * sCb;

  f32x8 acc[2][4];
#pragma unroll
  for (int i = 0; i < 2; ++i)
#pragma unroll
    for (int j = 0; j < 4; ++j)
#pragma unroll
      for (int e = 0; e < 8; ++e) acc[i][j][e] = 0.0f;

  if constexpr (TRANS_B) {
    // -------- TDM async double-buffered pipeline (K multiple of 64) --------
    const bf16* Abase = Ab + bm * (long)K;
    const bf16* Bbase = Bb + bn * (long)K;
    const unsigned ldsA0 = (unsigned)(size_t)&As[0][0][0];
    const unsigned ldsA1 = (unsigned)(size_t)&As[1][0][0];
    const unsigned ldsB0 = (unsigned)(size_t)&Bs[0][0][0];
    const unsigned ldsB1 = (unsigned)(size_t)&Bs[1][0][0];

    if (wid == 0) {                      // prologue: fill buffer 0
      tdm_load_tile(Abase, ldsA0, K);
      tdm_load_tile(Bbase, ldsB0, K);
    }
    for (int kk = 0; kk < K; kk += 64) {
      if (wid == 0) {                    // prefetch kk+32 into buffer 1
        tdm_load_tile(Abase + kk + 32, ldsA1, K);
        tdm_load_tile(Bbase + kk + 32, ldsB1, K);
        __builtin_amdgcn_s_wait_tensorcnt(2);   // buffer 0 done, buffer 1 in flight
      }
      __syncthreads();
      mma_step(As[0], Bs[0], acc, wr, wc, l15, lh);
      __syncthreads();
      if (wid == 0) {
        if (kk + 64 < K) {               // prefetch kk+64 into buffer 0
          tdm_load_tile(Abase + kk + 64, ldsA0, K);
          tdm_load_tile(Bbase + kk + 64, ldsB0, K);
          __builtin_amdgcn_s_wait_tensorcnt(2);
        } else {
          __builtin_amdgcn_s_wait_tensorcnt(0);
        }
      }
      __syncthreads();
      mma_step(As[1], Bs[1], acc, wr, wc, l15, lh);
      __syncthreads();
    }
  } else {
    // -------- sync path: B[K,N] row-major, transpose-staged into LDS --------
    const int arow  = tid >> 1;
    const int ahalf = (tid & 1) * 16;
    for (int kk = 0; kk < K; kk += 32) {
      {
        const bf16* s = Ab + (bm + arow) * (long)K + kk + ahalf;
        *(bf16x8*)&As[0][arow][ahalf]     = *(const bf16x8*)s;
        *(bf16x8*)&As[0][arow][ahalf + 8] = *(const bf16x8*)(s + 8);
      }
      {
        const int kr = tid >> 3;
        const int nc = (tid & 7) * 16;
        const bf16* s = Bb + (long)(kk + kr) * N + bn + nc;
        bf16x8 v0 = *(const bf16x8*)s;
        bf16x8 v1 = *(const bf16x8*)(s + 8);
#pragma unroll
        for (int i = 0; i < 8; ++i) Bs[0][nc + i][kr]     = v0[i];
#pragma unroll
        for (int i = 0; i < 8; ++i) Bs[0][nc + 8 + i][kr] = v1[i];
      }
      __syncthreads();
      mma_step(As[0], Bs[0], acc, wr, wc, l15, lh);
      __syncthreads();
    }
  }

  // epilogue: C/D layout: VGPR r -> M = 8*lh + r, N = l15
#pragma unroll
  for (int tm = 0; tm < 2; ++tm)
#pragma unroll
    for (int tn = 0; tn < 4; ++tn)
#pragma unroll
      for (int r = 0; r < 8; ++r) {
        const long m = bm + wr * 32 + tm * 16 + lh * 8 + r;
        const long n = bn + wc * 64 + tn * 16 + l15;
        float v = acc[tm][tn][r];
        if (BIAS)  v += bias[n];
        if (SILU)  v = v / (1.0f + __expf(-v));
        if (RESID) v += resid[cb + m * (long)N + n];
        const long idx = cb + m * (long)N + n;
        if (OUT_BF16) outB[idx] = f2bf(v);
        else          outF[idx] = v;
      }
}

// ---------- GroupNorm over (S, D/G) per (batch, group); writes bf16 ----------
__global__ __launch_bounds__(256)
void groupnorm_bf16(const float* __restrict__ x, const float* __restrict__ g,
                    const float* __restrict__ b, bf16* __restrict__ out,
                    int S, int D, int G)
{
  __shared__ float buf[256];
  const int bb = blockIdx.x / G;
  const int gg = blockIdx.x % G;
  const int cg = D / G;
  const long base = (long)bb * S * D + (long)gg * cg;
  const int n = S * cg;
  float s0 = 0.f, s1 = 0.f;
  for (int i = threadIdx.x; i < n; i += 256) {
    const int s = i / cg, c = i % cg;
    const float v = x[base + (long)s * D + c];
    s0 += v; s1 += v * v;
  }
  const float tot  = blk_sum(s0, buf);
  const float tot2 = blk_sum(s1, buf);
  const float mean = tot / (float)n;
  const float var  = tot2 / (float)n - mean * mean;
  const float rstd = rsqrtf(var + 1e-5f);
  for (int i = threadIdx.x; i < n; i += 256) {
    const int s = i / cg, c = i % cg;
    const long off = base + (long)s * D + c;
    out[off] = f2bf((x[off] - mean) * rstd * g[gg * cg + c] + b[gg * cg + c]);
  }
}

// ---------- row-wise L2 normalize, fp32 in -> bf16 out ----------
__global__ __launch_bounds__(256)
void l2norm_bf16(const float* __restrict__ in, bf16* __restrict__ out, int Dd)
{
  __shared__ float buf[256];
  const long row = blockIdx.x;
  const float* r = in + row * (long)Dd;
  float ss = 0.f;
  for (int i = threadIdx.x; i < Dd; i += 256) { const float v = r[i]; ss += v * v; }
  const float tot = blk_sum(ss, buf);
  const float inv = 1.0f / fmaxf(sqrtf(tot), 1e-12f);
  for (int i = threadIdx.x; i < Dd; i += 256) out[row * (long)Dd + i] = f2bf(r[i] * inv);
}

// ---------- row-wise softmax(scale * s), fp32 in -> bf16 out ----------
__global__ __launch_bounds__(256)
void softmax_bf16(const float* __restrict__ Sc, bf16* __restrict__ P, int L, float scale)
{
  __shared__ float buf[256];
  const long row = blockIdx.x;
  const float* r = Sc + row * (long)L;
  float mx = -3.0e38f;
  for (int i = threadIdx.x; i < L; i += 256) mx = fmaxf(mx, r[i] * scale);
  mx = blk_max(mx, buf);
  float sum = 0.f;
  for (int i = threadIdx.x; i < L; i += 256) sum += __expf(r[i] * scale - mx);
  sum = blk_sum(sum, buf);
  const float inv = 1.0f / sum;
  for (int i = threadIdx.x; i < L; i += 256)
    P[row * (long)L + i] = f2bf(__expf(r[i] * scale - mx) * inv);
}

// ---------- fp32 -> bf16 convert ----------
__global__ __launch_bounds__(256)
void cvt_bf16(const float* __restrict__ in, bf16* __restrict__ out, long n)
{
  const long i = (long)blockIdx.x * 256 + threadIdx.x;
  if (i < n) out[i] = f2bf(in[i]);
}

// ---------- host-side orchestration ----------
extern "C" void kernel_launch(void* const* d_in, const int* in_sizes, int n_in,
                              void* d_out, int out_size, void* d_ws, size_t ws_size,
                              hipStream_t stream)
{
  (void)in_sizes; (void)n_in; (void)out_size; (void)ws_size;
  const int Bb = 8, S = 2048, D = 1024, G = 16;
  const int BS = Bb * S; // 16384 rows

  const float* x    = (const float*)d_in[0];
  const float* gn1g = (const float*)d_in[1];
  const float* gn1b = (const float*)d_in[2];
  const float* wq   = (const float*)d_in[3];
  const float* wk   = (const float*)d_in[4];
  const float* wv   = (const float*)d_in[5];
  const float* wret = (const float*)d_in[6];
  const float* wout = (const float*)d_in[7];
  const float* bout = (const float*)d_in[8];
  const float* gn2g = (const float*)d_in[9];
  const float* gn2b = (const float*)d_in[10];
  const float* w1   = (const float*)d_in[11];
  const float* b1   = (const float*)d_in[12];
  const float* w2   = (const float*)d_in[13];
  const float* b2   = (const float*)d_in[14];
  float* out = (float*)d_out;

  char* ws = (char*)d_ws;
  const size_t MB = 1024ull * 1024ull;
  bf16*  h_bf   = (bf16*)(ws + 0);        // 32MB  (also reused as h2)
  bf16*  q_bf   = (bf16*)(ws + 32 * MB);  // 32MB  (also reused as y1)
  bf16*  k_bf   = (bf16*)(ws + 64 * MB);  // 32MB
  bf16*  v_bf   = (bf16*)(ws + 96 * MB);  // 32MB
  float* qpre   = (float*)(ws + 128 * MB);// 64MB \ overlap with scores (128MB)
  float* kpre   = (float*)(ws + 192 * MB);// 64MB /
  float* scores = (float*)(ws + 128 * MB);// 128MB (also reused as f_bf, 64MB)
  bf16*  f_bf   = (bf16*)(ws + 128 * MB);
  bf16*  P_bf   = (bf16*)(ws + 256 * MB); // 64MB
  bf16*  ao_bf  = (bf16*)(ws + 320 * MB); // 32MB
  float* x1     = (float*)(ws + 352 * MB);// 64MB
  bf16*  wq_bf   = (bf16*)(ws + 416 * MB);
  bf16*  wk_bf   = wq_bf   + (long)D * D;
  bf16*  wv_bf   = wk_bf   + (long)D * D;
  bf16*  wret_bf = wv_bf   + (long)D * D;
  bf16*  wout_bf = wret_bf + (long)D * D;
  bf16*  w1_bf   = wout_bf + (long)D * D;
  bf16*  w2_bf   = w1_bf   + 2l * D * D;
  bf16*  y1_bf   = q_bf;
  bf16*  h2_bf   = h_bf;

  const dim3 blk(256);
  const long DD = (long)D * D;
  cvt_bf16<<<dim3((unsigned)((DD + 255) / 256)),     blk, 0, stream>>>(wq,   wq_bf,   DD);
  cvt_bf16<<<dim3((unsigned)((DD + 255) / 256)),     blk, 0, stream>>>(wk,   wk_bf,   DD);
  cvt_bf16<<<dim3((unsigned)((DD + 255) / 256)),     blk, 0, stream>>>(wv,   wv_bf,   DD);
  cvt_bf16<<<dim3((unsigned)((DD + 255) / 256)),     blk, 0, stream>>>(wret, wret_bf, DD);
  cvt_bf16<<<dim3((unsigned)((DD + 255) / 256)),     blk, 0, stream>>>(wout, wout_bf, DD);
  cvt_bf16<<<dim3((unsigned)((2 * DD + 255) / 256)), blk, 0, stream>>>(w1,   w1_bf,   2 * DD);
  cvt_bf16<<<dim3((unsigned)((2 * DD + 255) / 256)), blk, 0, stream>>>(w2,   w2_bf,   2 * DD);

  // h = GroupNorm1(x) -> bf16
  groupnorm_bf16<<<dim3(Bb * G), blk, 0, stream>>>(x, gn1g, gn1b, h_bf, S, D, G);

  // q_pre / k_pre (fp32 for exact l2norm), v (bf16)
  gemm_wmma<true, false, false, false, false><<<dim3(D / 128, BS / 128, 1), blk, 0, stream>>>(
      h_bf, wq_bf, nullptr, nullptr, qpre, nullptr, BS, D, D, 0, 0, 0);
  gemm_wmma<true, false, false, false, false><<<dim3(D / 128, BS / 128, 1), blk, 0, stream>>>(
      h_bf, wk_bf, nullptr, nullptr, kpre, nullptr, BS, D, D, 0, 0, 0);
  gemm_wmma<true, false, false, false, true><<<dim3(D / 128, BS / 128, 1), blk, 0, stream>>>(
      h_bf, wv_bf, nullptr, nullptr, nullptr, v_bf, BS, D, D, 0, 0, 0);

  l2norm_bf16<<<dim3(BS), blk, 0, stream>>>(qpre, q_bf, D);
  l2norm_bf16<<<dim3(BS), blk, 0, stream>>>(kpre, k_bf, D);

  // scores[b] = q[b] @ k[b]^T  (fp32)
  gemm_wmma<true, false, false, false, false><<<dim3(S / 128, S / 128, Bb), blk, 0, stream>>>(
      q_bf, k_bf, nullptr, nullptr, scores, nullptr, S, S, D,
      (long)S * D, (long)S * D, (long)S * S);

  // P = softmax(scores / 32) -> bf16
  softmax_bf16<<<dim3(BS), blk, 0, stream>>>(scores, P_bf, S, 0.03125f);

  // ao[b] = P[b] @ V[b]   (B not transposed -> sync path)
  gemm_wmma<false, false, false, false, true><<<dim3(D / 128, S / 128, Bb), blk, 0, stream>>>(
      P_bf, v_bf, nullptr, nullptr, nullptr, ao_bf, S, D, S,
      (long)S * S, (long)S * D, (long)S * D);

  // y1 = ao @ w_ret^T
  gemm_wmma<true, false, false, false, true><<<dim3(D / 128, BS / 128, 1), blk, 0, stream>>>(
      ao_bf, wret_bf, nullptr, nullptr, nullptr, y1_bf, BS, D, D, 0, 0, 0);

  // x1 = y1 @ w_out^T + b_out + x
  gemm_wmma<true, true, true, false, false><<<dim3(D / 128, BS / 128, 1), blk, 0, stream>>>(
      y1_bf, wout_bf, bout, x, x1, nullptr, BS, D, D, 0, 0, 0);

  // h2 = GroupNorm2(x1) -> bf16
  groupnorm_bf16<<<dim3(Bb * G), blk, 0, stream>>>(x1, gn2g, gn2b, h2_bf, S, D, G);

  // f = silu(h2 @ w1^T + b1) -> bf16
  gemm_wmma<true, true, false, true, true><<<dim3(2 * D / 128, BS / 128, 1), blk, 0, stream>>>(
      h2_bf, w1_bf, b1, nullptr, nullptr, f_bf, BS, 2 * D, D, 0, 0, 0);

  // out = f @ w2^T + b2 + x1
  gemm_wmma<true, true, true, false, false><<<dim3(D / 128, BS / 128, 1), blk, 0, stream>>>(
      f_bf, w2_bf, b2, x1, out, nullptr, BS, D, 2 * D, 0, 0, 0);
}
